// InterAttention_45990509806289
// MI455X (gfx1250) — matrix operand order repaired and checked by reading, not compiled, not run on hardware
//
#include <hip/hip_runtime.h>

typedef _Float16 half_t;
typedef _Float16 v16h __attribute__((ext_vector_type(16)));
typedef _Float16 v8h  __attribute__((ext_vector_type(8)));
typedef float    v8f  __attribute__((ext_vector_type(8)));

#define B_   64
#define L1_  512
#define L2_  512
#define D_   256
#define NT1_ (B_ * L1_)        // 32768 rows for tower 1
#define NT2_ (B_ * L2_)        // 32768 rows for tower 2
#define NT_  (NT1_ + NT2_)     // 65536 total token rows

// ---------------- WMMA fragment loaders (row-major [*, K] operands) ----------------
// A (16x32 f16): lane&15 = M row; lanes<16 hold K 0..15 contiguous, lanes>=16 hold K 16..31 contiguous
__device__ __forceinline__ v16h load_frag_a(const half_t* p0, int ld, int lane) {
  const int r  = lane & 15;
  const int kk = (lane & 16) ? 16 : 0;
  const half_t* p = p0 + (long)r * ld + kk;
  v8h lo = *(const v8h*)(p);        // K kk..kk+7   -> elements 0..7  (VGPR 0..3)
  v8h hi = *(const v8h*)(p + 8);    // K kk+8..+15  -> elements 8..15 (VGPR 4..7)
  return __builtin_shufflevector(lo, hi, 0,1,2,3,4,5,6,7,8,9,10,11,12,13,14,15);
}
// B (32x16 f16) loaded from Bt row-major [N, K]: lane&15 = N column;
// lanes<16 hold K 0..15 contiguous, lanes>=16 hold K 16..31 contiguous.
__device__ __forceinline__ v16h load_frag_b(const half_t* p0, int ld, int lane) {
  const int n  = lane & 15;
  const int kk = (lane & 16) ? 16 : 0;
  return *(const v16h*)(p0 + (long)n * ld + kk);
}

__device__ __forceinline__ float leaky01(float x) { return x > 0.f ? x : 0.01f * x; }

// ---------------- Generic batched WMMA GEMM ----------------
// C[b] = act( A[b] @ Bt[b]^T + bias ) * rowmask      (Bt is row-major [N, K])
// flags: bit0 = leaky relu, bit1 = f16 output (else f32)
// Block: 256 thr = 8 waves as 4(M) x 2(N); wave tile 32x64 = 8 WMMA accums; block tile 128x128.
// Per K-step: 2 A frags + 4 B frags feed 8 v_wmma -> ~21 flops per loaded byte per lane.
__global__ __launch_bounds__(256)
void gemm_wmma_kernel(const half_t* __restrict__ A, int lda, long sA,
                      const half_t* __restrict__ Bt, int ldb, long sB,
                      const float* __restrict__ bias,
                      const float* __restrict__ rowmask,
                      void* __restrict__ Cout, int ldc, long sC,
                      int M, int N, int K, int flags) {
  const int lane = threadIdx.x & 31;
  const int wave = threadIdx.x >> 5;
  const int wm = wave & 3, wn = wave >> 2;
  const int m0 = blockIdx.y * 128 + wm * 32;
  const int n0 = blockIdx.x * 128 + wn * 64;
  if (m0 >= M || n0 >= N) return;
  const int b = blockIdx.z;
  A  += (long)b * sA;
  Bt += (long)b * sB;

  const half_t* Apf = A + (long)(m0 + (lane & 15)) * lda;   // per-lane prefetch row

  v8f acc[2][4] = {};
  for (int k0 = 0; k0 < K; k0 += 32) {
    v16h a0 = load_frag_a(A + (long)m0 * lda + k0,        lda, lane);
    v16h a1 = load_frag_a(A + (long)(m0 + 16) * lda + k0, lda, lane);
    v16h bf[4];
#pragma unroll
    for (int j = 0; j < 4; ++j)
      bf[j] = load_frag_b(Bt + (long)(n0 + 16 * j) * ldb + k0, ldb, lane);
    // Pull the streaming A operand two K-steps ahead (HBM -> caches).
    if (k0 + 64 < K) __builtin_prefetch(Apf + k0 + 64, 0, 0);
#pragma unroll
    for (int j = 0; j < 4; ++j) {
      acc[0][j] = __builtin_amdgcn_wmma_f32_16x16x32_f16(false, a0, false, bf[j], (short)0, acc[0][j], false, false);
      acc[1][j] = __builtin_amdgcn_wmma_f32_16x16x32_f16(false, a1, false, bf[j], (short)0, acc[1][j], false, false);
    }
  }

  const bool do_leaky = (flags & 1) != 0;
  const bool out_f16  = (flags & 2) != 0;
  float*  cf = (float*)Cout  + (long)b * sC;
  half_t* ch = (half_t*)Cout + (long)b * sC;

  auto store_tile = [&](v8f c, int mm, int nn) {
    const int col = nn + (lane & 15);
    const float bv = bias ? bias[col] : 0.f;
    const int rbase = mm + ((lane & 16) ? 8 : 0);
#pragma unroll
    for (int r = 0; r < 8; ++r) {
      const int row = rbase + r;
      float v = c[r] + bv;
      if (do_leaky) v = leaky01(v);
      if (rowmask)  v *= rowmask[row];
      if (out_f16) ch[(long)row * ldc + col] = (half_t)v;
      else         cf[(long)row * ldc + col] = v;
    }
  };
#pragma unroll
  for (int i = 0; i < 2; ++i)
#pragma unroll
    for (int j = 0; j < 4; ++j)
      store_tile(acc[i][j], m0 + 16 * i, n0 + 16 * j);
}

// ---------------- Elementwise helpers ----------------
__global__ void f32_to_f16_kernel(const float* __restrict__ in, half_t* __restrict__ out, int n) {
  int i = blockIdx.x * 256 + threadIdx.x;
  if (i < n) out[i] = (half_t)in[i];
}
__global__ void i32_to_f32_kernel(const int* __restrict__ in, float* __restrict__ out, int n) {
  int i = blockIdx.x * 256 + threadIdx.x;
  if (i < n) out[i] = (float)in[i];
}
// W [K,N] f32 row-major -> Wt [N,K] f16 row-major
__global__ void wtranspose_kernel(const float* __restrict__ W, half_t* __restrict__ Wt, int K, int N) {
  int i = blockIdx.x * 256 + threadIdx.x;
  if (i < N * K) {
    int n = i / K, k = i % K;
    Wt[i] = (half_t)W[(long)k * N + n];
  }
}

// ---------------- f16 transpose (32x32 LDS tiles), batched via grid.z ----------------
__global__ __launch_bounds__(256)
void transpose_f16_kernel(const half_t* __restrict__ src, int sld, long sstride,
                          half_t* __restrict__ dst, int dld, long dstride) {
  __shared__ half_t tile[32][33];
  src += (long)blockIdx.z * sstride;
  dst += (long)blockIdx.z * dstride;
  const int c0 = blockIdx.x * 32, r0 = blockIdx.y * 32;
  const int tx = threadIdx.x & 31, ty = threadIdx.x >> 5; // 32 x 8
#pragma unroll
  for (int i = 0; i < 32; i += 8)
    tile[ty + i][tx] = src[(long)(r0 + ty + i) * sld + (c0 + tx)];
  __syncthreads();
#pragma unroll
  for (int i = 0; i < 32; i += 8)
    dst[(long)(c0 + ty + i) * dld + (r0 + tx)] = tile[tx][ty + i];
}

// ---------------- Masked softmax over rows (axis 2): o1 = softmax(o,2)*vf, f16 out ----------------
__global__ __launch_bounds__(256)
void softmax_row_kernel(const float* __restrict__ o, const int* __restrict__ m1,
                        const int* __restrict__ m2, half_t* __restrict__ o1) {
  const int bl = blockIdx.x;            // b*L1 + l
  const int b = bl >> 9;
  const int t = threadIdx.x;
  half_t* out = o1 + (long)bl * L2_;
  if (m1[bl] == 0) {
    for (int m = t; m < L2_; m += 256) out[m] = (half_t)0.f;
    return;
  }
  const float* row = o + (long)bl * L2_;
  const int* m2b = m2 + b * L2_;
  __shared__ float red[256];
  float mx = -3.0e38f;
  for (int m = t; m < L2_; m += 256) if (m2b[m]) mx = fmaxf(mx, row[m]);
  red[t] = mx; __syncthreads();
  for (int s = 128; s; s >>= 1) { if (t < s) red[t] = fmaxf(red[t], red[t + s]); __syncthreads(); }
  mx = red[0]; __syncthreads();
  float sum = 0.f;
  for (int m = t; m < L2_; m += 256) if (m2b[m]) sum += __expf(row[m] - mx);
  red[t] = sum; __syncthreads();
  for (int s = 128; s; s >>= 1) { if (t < s) red[t] += red[t + s]; __syncthreads(); }
  const float inv = red[0] > 0.f ? 1.f / red[0] : 0.f;
  for (int m = t; m < L2_; m += 256)
    out[m] = m2b[m] ? (half_t)(__expf(row[m] - mx) * inv) : (half_t)0.f;
}

// ---------------- Masked softmax over cols (axis 1), written TRANSPOSED: o2T [B,L2,L1] f16 ----------------
__global__ __launch_bounds__(256)
void softmax_col_kernel(const float* __restrict__ o, const int* __restrict__ m1,
                        const int* __restrict__ m2, half_t* __restrict__ o2T) {
  const int bm = blockIdx.x;            // b*L2 + m
  const int b = bm >> 9, m = bm & 511;
  const int t = threadIdx.x;
  half_t* out = o2T + (long)bm * L1_;
  if (m2[bm] == 0) {
    for (int l = t; l < L1_; l += 256) out[l] = (half_t)0.f;
    return;
  }
  const float* col = o + (long)b * L1_ * L2_ + m;
  const int* m1b = m1 + b * L1_;
  __shared__ float red[256];
  float mx = -3.0e38f;
  for (int l = t; l < L1_; l += 256) if (m1b[l]) mx = fmaxf(mx, col[(long)l * L2_]);
  red[t] = mx; __syncthreads();
  for (int s = 128; s; s >>= 1) { if (t < s) red[t] = fmaxf(red[t], red[t + s]); __syncthreads(); }
  mx = red[0]; __syncthreads();
  float sum = 0.f;
  for (int l = t; l < L1_; l += 256) if (m1b[l]) sum += __expf(col[(long)l * L2_] - mx);
  red[t] = sum; __syncthreads();
  for (int s = 128; s; s >>= 1) { if (t < s) red[t] += red[t + s]; __syncthreads(); }
  const float inv = red[0] > 0.f ? 1.f / red[0] : 0.f;
  for (int l = t; l < L1_; l += 256)
    out[l] = m1b[l] ? (half_t)(__expf(col[(long)l * L2_] - mx) * inv) : (half_t)0.f;
}

// ---------------- Masked mean pooling: out[b,d] = sum_l cmp[row,d]*m[row] / sum_l m[row] ----------------
__global__ __launch_bounds__(256)
void pool_kernel(const float* __restrict__ cmp, const float* __restrict__ rowmask,
                 float* __restrict__ out, long rowoff, long outoff, int L) {
  const int b = blockIdx.x, d = threadIdx.x;        // D_ == 256 threads
  const float* base = cmp + (rowoff + (long)b * L) * D_;
  const float* mk = rowmask + rowoff + (long)b * L;
  float s = 0.f, cnt = 0.f;
  for (int l = 0; l < L; ++l) {
    const float mv = mk[l];
    cnt += mv;
    s += base[(long)l * D_ + d] * mv;
  }
  out[outoff + (long)b * D_ + d] = s / cnt;
}

extern "C" void kernel_launch(void* const* d_in, const int* in_sizes, int n_in,
                              void* d_out, int out_size, void* d_ws, size_t ws_size,
                              hipStream_t stream) {
  (void)in_sizes; (void)n_in; (void)out_size; (void)ws_size;
  const float* r1    = (const float*)d_in[0];
  const float* r2    = (const float*)d_in[1];
  const int*   mask1 = (const int*)  d_in[2];
  const int*   mask2 = (const int*)  d_in[3];
  const float* W1    = (const float*)d_in[4];
  const float* b1    = (const float*)d_in[5];
  const float* W2    = (const float*)d_in[6];
  const float* b2    = (const float*)d_in[7];
  const float* Wc1   = (const float*)d_in[8];
  const float* bc1   = (const float*)d_in[9];
  const float* Wc2   = (const float*)d_in[10];
  const float* bc2   = (const float*)d_in[11];
  float* out = (float*)d_out;

  // ---- workspace layout (bytes) ----
  char* w = (char*)d_ws;
  const size_t OFF_X    = 0;                         // xf16 [NT,256] f16  (reused as u)
  const size_t OFF_T    = OFF_X    + (size_t)NT_ * D_ * 2;           // t  [NT,256] f16
  const size_t OFF_CAT  = OFF_T    + (size_t)NT_ * D_ * 2;           // cat[NT,512] f16 (h | r_c)
  const size_t OFF_H1T  = OFF_CAT  + (size_t)NT_ * 2 * D_ * 2;       // h1T [B,256,512] f16
  const size_t OFF_H2T  = OFF_H1T  + (size_t)B_ * D_ * L1_ * 2;      // h2T [B,256,512] f16
  const size_t OFF_SC   = OFF_H2T  + (size_t)B_ * D_ * L2_ * 2;      // scores [B,512,512] f32 (reused as cmp f32)
  const size_t OFF_O1   = OFF_SC   + (size_t)B_ * L1_ * L2_ * 4;     // o1  [B,512,512] f16
  const size_t OFF_O2T  = OFF_O1   + (size_t)B_ * L1_ * L2_ * 2;     // o2T [B,512,512] f16
  const size_t OFF_RM   = OFF_O2T  + (size_t)B_ * L2_ * L1_ * 2;     // rowmask [NT] f32
  const size_t OFF_W1T  = OFF_RM   + (size_t)NT_ * 4;
  const size_t OFF_W2T  = OFF_W1T  + (size_t)D_ * D_ * 2;
  const size_t OFF_WC1T = OFF_W2T  + (size_t)D_ * D_ * 2;
  const size_t OFF_WC2T = OFF_WC1T + (size_t)D_ * 2 * D_ * 2;

  half_t* xf16 = (half_t*)(w + OFF_X);
  half_t* tf16 = (half_t*)(w + OFF_T);
  half_t* cat  = (half_t*)(w + OFF_CAT);
  half_t* h1T  = (half_t*)(w + OFF_H1T);
  half_t* h2T  = (half_t*)(w + OFF_H2T);
  float*  sc   = (float*) (w + OFF_SC);
  float*  cmp  = sc;                       // reuse
  half_t* o1   = (half_t*)(w + OFF_O1);
  half_t* o2T  = (half_t*)(w + OFF_O2T);
  float*  rm   = (float*) (w + OFF_RM);
  half_t* W1t  = (half_t*)(w + OFF_W1T);
  half_t* W2t  = (half_t*)(w + OFF_W2T);
  half_t* Wc1t = (half_t*)(w + OFF_WC1T);
  half_t* Wc2t = (half_t*)(w + OFF_WC2T);
  half_t* u    = xf16;                     // reuse

  // 1) convert inputs / masks / weights
  f32_to_f16_kernel<<<(NT1_ * D_ + 255) / 256, 256, 0, stream>>>(r1, xf16, NT1_ * D_);
  f32_to_f16_kernel<<<(NT2_ * D_ + 255) / 256, 256, 0, stream>>>(r2, xf16 + (size_t)NT1_ * D_, NT2_ * D_);
  i32_to_f32_kernel<<<(NT1_ + 255) / 256, 256, 0, stream>>>(mask1, rm, NT1_);
  i32_to_f32_kernel<<<(NT2_ + 255) / 256, 256, 0, stream>>>(mask2, rm + NT1_, NT2_);
  wtranspose_kernel<<<(D_ * D_ + 255) / 256, 256, 0, stream>>>(W1, W1t, D_, D_);
  wtranspose_kernel<<<(D_ * D_ + 255) / 256, 256, 0, stream>>>(W2, W2t, D_, D_);
  wtranspose_kernel<<<(D_ * 2 * D_ + 255) / 256, 256, 0, stream>>>(Wc1, Wc1t, 2 * D_, D_);
  wtranspose_kernel<<<(D_ * D_ + 255) / 256, 256, 0, stream>>>(Wc2, Wc2t, D_, D_);

  // 2) shared MLP: t = leaky(x@W1+b1); h = leaky(t@W2+b2)*m  -> cat[:,0:256]
  gemm_wmma_kernel<<<dim3(D_ / 128, NT_ / 128, 1), 256, 0, stream>>>(
      xf16, D_, 0, W1t, D_, 0, b1, nullptr, tf16, D_, 0, NT_, D_, D_, /*leaky|f16*/3);
  gemm_wmma_kernel<<<dim3(D_ / 128, NT_ / 128, 1), 256, 0, stream>>>(
      tf16, D_, 0, W2t, D_, 0, b2, rm, cat, 2 * D_, 0, NT_, D_, D_, 3);

  // 3) h1T, h2T transposes (for context GEMMs' B operands)
  transpose_f16_kernel<<<dim3(D_ / 32, L1_ / 32, B_), 256, 0, stream>>>(
      cat, 2 * D_, (long)L1_ * 2 * D_, h1T, L1_, (long)D_ * L1_);
  transpose_f16_kernel<<<dim3(D_ / 32, L2_ / 32, B_), 256, 0, stream>>>(
      cat + (size_t)NT1_ * 2 * D_, 2 * D_, (long)L2_ * 2 * D_, h2T, L2_, (long)D_ * L2_);

  // 4) scores: o[b] = h1[b] @ h2[b]^T  (f32)
  gemm_wmma_kernel<<<dim3(L2_ / 128, L1_ / 128, B_), 256, 0, stream>>>(
      cat, 2 * D_, (long)L1_ * 2 * D_,
      cat + (size_t)NT1_ * 2 * D_, 2 * D_, (long)L2_ * 2 * D_,
      nullptr, nullptr, sc, L2_, (long)L1_ * L2_, L1_, L2_, D_, 0);

  // 5) masked softmaxes
  softmax_row_kernel<<<NT1_, 256, 0, stream>>>(sc, mask1, mask2, o1);
  softmax_col_kernel<<<NT2_, 256, 0, stream>>>(sc, mask1, mask2, o2T);

  // 6) r1_c = o1 @ h2 -> cat1[:,256:512];  r2_c = o2^T @ h1 -> cat2[:,256:512]
  gemm_wmma_kernel<<<dim3(D_ / 128, L1_ / 128, B_), 256, 0, stream>>>(
      o1, L2_, (long)L1_ * L2_, h2T, L2_, (long)D_ * L2_,
      nullptr, nullptr, cat + D_, 2 * D_, (long)L1_ * 2 * D_, L1_, D_, L2_, /*f16*/2);
  gemm_wmma_kernel<<<dim3(D_ / 128, L2_ / 128, B_), 256, 0, stream>>>(
      o2T, L1_, (long)L2_ * L1_, h1T, L1_, (long)D_ * L1_,
      nullptr, nullptr, cat + (size_t)NT1_ * 2 * D_ + D_, 2 * D_, (long)L2_ * 2 * D_, L2_, D_, L1_, 2);

  // 7) compare MLP: u = leaky(cat@Wc1+bc1); cmp = leaky(u@Wc2+bc2) (f32)
  gemm_wmma_kernel<<<dim3(D_ / 128, NT_ / 128, 1), 256, 0, stream>>>(
      cat, 2 * D_, 0, Wc1t, 2 * D_, 0, bc1, nullptr, u, D_, 0, NT_, D_, 2 * D_, 3);
  gemm_wmma_kernel<<<dim3(D_ / 128, NT_ / 128, 1), 256, 0, stream>>>(
      u, D_, 0, Wc2t, D_, 0, bc2, nullptr, cmp, D_, 0, NT_, D_, D_, /*leaky,f32*/1);

  // 8) masked mean pooling -> (r1_pool, r2_pool) concatenated
  pool_kernel<<<B_, 256, 0, stream>>>(cmp, rm, out, 0, 0, L1_);
  pool_kernel<<<B_, 256, 0, stream>>>(cmp, rm, out, NT1_, (long)B_ * D_, L2_);
}
